// ODE_RNN_8864812499528
// MI455X (gfx1250) — matrix operand order repaired and checked by reading, not compiled, not run on hardware
//
#include <hip/hip_runtime.h>

// ---------------------------------------------------------------------------
// ODE-RNN (latent ODE w/ Dopri5 + GRU) for gfx1250.
//   B=256, T=64, OB=32, AC=16, L=128, H=256, gates=384, NSUB=2.
// Persistent sequential kernel: 16 workgroups (one 16-row batch tile each),
// 512 threads = 16 wave32s covering the N dimension with
// v_wmma_f32_16x16x32_bf16. dyn_w0+dyn_w1 (192KB bf16) are staged into LDS
// once per block via the Tensor Data Mover (tensor_load_to_lds +
// s_wait_tensorcnt). Decoder runs as a separate fully parallel kernel
// (1024 tiles). Weights are converted to bf16 into d_ws each launch.
// Workspace need: (405504 + 64*256*128) bf16 elems = 5,005,312 bytes.
// ---------------------------------------------------------------------------

#define BDIM 256
#define TDIM 64
#define OBD  32
#define ACD  16
#define LDIM 128
#define HDIM 256
#define GDIM 384
#define MT   16
#define NWAVES 16
#define NTHREADS 512

typedef __attribute__((ext_vector_type(16))) __bf16          bf16x16;
typedef __attribute__((ext_vector_type(8)))  float           f32x8;
typedef __attribute__((ext_vector_type(8)))  unsigned short  u16x8;
typedef __attribute__((ext_vector_type(16))) unsigned short  u16x16;
typedef __attribute__((ext_vector_type(4)))  unsigned int    u32x4;
typedef __attribute__((ext_vector_type(8)))  int             i32x8;
typedef __attribute__((ext_vector_type(4)))  int             i32x4;

union BFU { u16x16 u; bf16x16 b; };

// Dormand-Prince tableau
__constant__ float cDA[5][5] = {
  {0.2f, 0.f, 0.f, 0.f, 0.f},
  {3.f/40.f, 9.f/40.f, 0.f, 0.f, 0.f},
  {44.f/45.f, -56.f/15.f, 32.f/9.f, 0.f, 0.f},
  {19372.f/6561.f, -25360.f/2187.f, 64448.f/6561.f, -212.f/729.f, 0.f},
  {9017.f/3168.f, -355.f/33.f, 46732.f/5247.f, 49.f/176.f, -5103.f/18656.f}};
__constant__ float cDB[6] = {35.f/384.f, 0.f, 500.f/1113.f, 125.f/192.f,
                             -2187.f/6784.f, 11.f/84.f};

// bf16 weight workspace layout (ushort element offsets, all 64-elem aligned)
#define OFF_ENC_W0 0L        // [256,32]
#define OFF_ENC_W1 8192L     // [256,256]
#define OFF_ENC_W2 73728L    // [128,256]
#define OFF_DYN_W0 106496L   // [256,128]   } contiguous 98304 elems (192KB):
#define OFF_DYN_W1 139264L   // [256,256]   } TDM-preloaded to LDS
#define OFF_DYN_W2 204800L   // [128,256]
#define OFF_DEC_W0 237568L   // [256,128]
#define OFF_DEC_W1 270336L   // [256,256]
#define OFF_DEC_W2 335872L   // [32,256]
#define OFF_WIH    344064L   // [384,32]  (K padded 16->32 with zeros)
#define OFF_WHH    356352L   // [384,128]
#define OFF_LAT    405504L   // [T, B, L] bf16 latents

#define WLDS_ELEMS 98304     // dyn_w0 (32768) + dyn_w1 (65536) in LDS

// round-to-nearest-even f32 -> bf16 (as raw ushort)
__device__ __forceinline__ unsigned short f2bf(float f) {
  union { float f; unsigned int u; } c; c.f = f;
  unsigned int r = c.u + 0x7FFFu + ((c.u >> 16) & 1u);
  return (unsigned short)(r >> 16);
}

// A fragment (16x32, MxK) from a bf16 (ushort) row-major buffer, row stride lda.
// Lane l: m = l&15, half = l>>4; elems 0..7 <- k0+half*8+i, 8..15 <- k0+16+half*8+i.
__device__ __forceinline__ bf16x16 a_frag_lds(const unsigned short* A, int lda,
                                              int k0, int lane) {
  int m = lane & 15, hf = lane >> 4;
  const unsigned short* p = A + m * lda + k0 + hf * 8;
  u16x8 lo = *(const u16x8*)(p);
  u16x8 hi = *(const u16x8*)(p + 16);
  BFU r;
#pragma unroll
  for (int i = 0; i < 8; ++i) { r.u[i] = lo[i]; r.u[8 + i] = hi[i]; }
  return r.b;
}

// A fragment from f32 row-major source, single K-tile (k in [0,32)), zero-pad k>=kmax.
__device__ __forceinline__ bf16x16 a_frag_f32(const float* A, int lda, int kmax,
                                              int lane) {
  int m = lane & 15, hf = lane >> 4;
  const float* p = A + (long)m * lda;
  BFU r;
#pragma unroll
  for (int i = 0; i < 8; ++i) {
    int ka = hf * 8 + i;
    int kb = 16 + hf * 8 + i;
    r.u[i]     = (ka < kmax) ? f2bf(p[ka]) : (unsigned short)0;
    r.u[8 + i] = (kb < kmax) ? f2bf(p[kb]) : (unsigned short)0;
  }
  return r.b;
}

// B fragment (32x16, KxN) from bf16 weight W [N,K] row-major (computes x @ W.T):
// lane l holds weight row n0+(l&15); elems 0..7 <- k0+half*8+i, 8..15 <- k0+16+half*8+i.
__device__ __forceinline__ bf16x16 b_frag(const unsigned short* W, int ldw,
                                          int n0, int k0, int lane) {
  int n = n0 + (lane & 15), hf = lane >> 4;
  const unsigned short* p = W + (long)n * ldw + k0 + hf * 8;
  u16x8 lo = *(const u16x8*)(p);
  u16x8 hi = *(const u16x8*)(p + 16);
  BFU r;
#pragma unroll
  for (int i = 0; i < 8; ++i) { r.u[i] = lo[i]; r.u[8 + i] = hi[i]; }
  return r.b;
}

__device__ __forceinline__ f32x8 wmma_bf16(bf16x16 a, bf16x16 b, f32x8 c) {
  return __builtin_amdgcn_wmma_f32_16x16x32_bf16(false, a, false, b, (short)0, c,
                                                 false, false);
}

// Epilogue: D tile -> bf16 mirror and/or f32 buffer (+bias, optional relu).
__device__ __forceinline__ void store_tile(f32x8 acc, const float* bias, int n0,
                                           int lane, unsigned short* Db, float* Df,
                                           int ldd, bool relu) {
  int nl = lane & 15, hf = lane >> 4;
  float bv = bias ? bias[n0 + nl] : 0.f;
#pragma unroll
  for (int r = 0; r < 8; ++r) {
    int m = hf * 8 + r;
    float v = acc[r] + bv;
    if (relu) v = fmaxf(v, 0.f);
    if (Db) Db[m * ldd + n0 + nl] = f2bf(v);
    if (Df) Df[m * ldd + n0 + nl] = v;
  }
}

// Dense layer: out[16,N] = act[16,K] @ W[N,K]^T (+bias, relu). 16 waves over N.
__device__ __forceinline__ void layer_lds(const unsigned short* A, int lda,
                                          const unsigned short* W, int K,
                                          const float* bias, int N,
                                          unsigned short* Db, float* Df, bool relu,
                                          int wave, int lane) {
  for (int n0 = wave * 16; n0 < N; n0 += NWAVES * 16) {
    f32x8 acc = {};
    for (int k0 = 0; k0 < K; k0 += 32)
      acc = wmma_bf16(a_frag_lds(A, lda, k0, lane), b_frag(W, K, n0, k0, lane), acc);
    store_tile(acc, bias, n0, lane, Db, Df, N, relu);
  }
}

// Dense layer with f32 global A, single K-tile (K<=32, W has ldw=32).
__device__ __forceinline__ void layer_f32A(const float* A, int lda, int kmax,
                                           const unsigned short* W, const float* bias,
                                           int N, unsigned short* Db, float* Df,
                                           bool relu, int wave, int lane) {
  bf16x16 a = a_frag_f32(A, lda, kmax, lane);
  for (int n0 = wave * 16; n0 < N; n0 += NWAVES * 16) {
    f32x8 acc = {};
    acc = wmma_bf16(a, b_frag(W, 32, n0, 0, lane), acc);
    store_tile(acc, bias, n0, lane, Db, Df, N, relu);
  }
}

// dynamics f(y): L -> H -> H -> L (relu, relu, identity) -> kout (f32)
__device__ __forceinline__ void eval_f(const unsigned short* in,
                                       const unsigned short* w0,
                                       const unsigned short* w1,
                                       const unsigned short* w2,
                                       const float* b0, const float* b1,
                                       const float* b2,
                                       unsigned short (*h1)[HDIM],
                                       unsigned short (*h2)[HDIM],
                                       float* kout, int wave, int lane) {
  layer_lds(in, LDIM, w0, LDIM, b0, HDIM, &h1[0][0], nullptr, true, wave, lane);
  __syncthreads();
  layer_lds(&h1[0][0], HDIM, w1, HDIM, b1, HDIM, &h2[0][0], nullptr, true, wave, lane);
  __syncthreads();
  layer_lds(&h2[0][0], HDIM, w2, HDIM, b2, LDIM, nullptr, kout, false, wave, lane);
  __syncthreads();
}

// GRU pointwise: r,z = sigmoid; n = tanh(gin + r*(ghn + bn)); y = (1-z)*n + z*y
__device__ __forceinline__ void gru_apply(const float* gi, const float* gh,
                                          const float* bn, float (*yfv)[LDIM],
                                          unsigned short (*ybv)[LDIM],
                                          unsigned short* latb, int t, int b0,
                                          int tid) {
  for (int idx = tid; idx < MT * LDIM; idx += NTHREADS) {
    int m = idx >> 7, j = idx & (LDIM - 1);
    float gir = gi[m * GDIM + j];
    float giz = gi[m * GDIM + LDIM + j];
    float gin = gi[m * GDIM + 2 * LDIM + j];
    float ghr = gh[m * GDIM + j];
    float ghz = gh[m * GDIM + LDIM + j];
    float ghn = gh[m * GDIM + 2 * LDIM + j];
    float r = 1.f / (1.f + __expf(-(gir + ghr)));
    float z = 1.f / (1.f + __expf(-(giz + ghz)));
    float n = tanhf(gin + r * (ghn + bn[j]));
    float y = (1.f - z) * n + z * yfv[m][j];
    yfv[m][j] = y;
    unsigned short hb = f2bf(y);
    ybv[m][j] = hb;
    latb[((long)t * BDIM + b0 + m) * LDIM + j] = hb;
  }
}

// TDM: copy `units8` 8-byte units from global `src` to LDS `dstLds` (1-D tile).
// D# per CDNA5 ISA. Issued by the calling wave.
// This toolchain has the 6-arg builtin: (u32x4, i32x8, i32x4, i32x4, i32x8, cpol).
__device__ __forceinline__ void tdm_load_1d(const void* src, void* dstLds,
                                            unsigned units8) {
  unsigned long long ga = (unsigned long long)src;
  unsigned lds = (unsigned)(unsigned long long)dstLds;  // flat addr low 32 = LDS offset
  u32x4 g0;
  g0[0] = 1u;                                            // count=1, user D#
  g0[1] = lds;                                           // lds_addr
  g0[2] = (unsigned)(ga & 0xFFFFFFFFull);                // global_addr[31:0]
  g0[3] = (unsigned)((ga >> 32) & 0x1FFFFFFull) | (2u << 30);  // addr[56:32] | type=2
  i32x8 g1;
  g1[0] = 3 << 16;                                       // data_size = 8B
  g1[1] = (int)((units8 & 0xFFFFu) << 16);               // tensor_dim0[15:0]
  g1[2] = (int)(units8 >> 16);                           // tensor_dim0[31:16]
  g1[3] = (int)((units8 & 0xFFFFu) << 16);               // tile_dim0 (16-bit)
  g1[4] = 0;                                             // tile_dim1/2 = 0 (1-D)
  g1[5] = (int)units8;                                   // tensor_dim0_stride[31:0]
  g1[6] = 0;
  g1[7] = 0;
  i32x4 z4 = {0, 0, 0, 0};
  i32x8 z8 = {0, 0, 0, 0, 0, 0, 0, 0};
  __builtin_amdgcn_tensor_load_to_lds(g0, g1, z4, z4, z8, 0);
}

// ---------------------------------------------------------------------------
// Main sequential kernel: 16 blocks x 512 threads; block = one 16-row batch tile.
// ---------------------------------------------------------------------------
__global__ __launch_bounds__(NTHREADS) void ode_rnn_seq(
    const float* __restrict__ ob, const float* __restrict__ acs,
    const float* __restrict__ times, unsigned short* __restrict__ wsb,
    const float* __restrict__ enc_b0, const float* __restrict__ enc_b1,
    const float* __restrict__ enc_b2, const float* __restrict__ dyn_b0,
    const float* __restrict__ dyn_b1, const float* __restrict__ dyn_b2,
    const float* __restrict__ gru_bih, const float* __restrict__ gru_bn) {
  __shared__ __align__(128) unsigned short wlds[WLDS_ELEMS];  // dyn w0 | w1 (192KB)
  __shared__ __align__(32) unsigned short yb[MT][LDIM];    // latent bf16
  __shared__ __align__(32) unsigned short xb[MT][LDIM];    // dopri stage input bf16
  __shared__ __align__(32) unsigned short h1b[MT][HDIM];
  __shared__ __align__(32) unsigned short h2b[MT][HDIM];
  __shared__ float yf[MT][LDIM];                           // latent f32
  __shared__ union Scr {                                   // 48KB, phase-exclusive
    float kf[6][MT][LDIM];   // dopri stages
    float g[2][MT][GDIM];    // gru gates: gi, gh
  } scr;
  __shared__ float hstep[MT];

  const int b0 = blockIdx.x * MT;
  const int tid = threadIdx.x, lane = tid & 31, wave = tid >> 5;

  const unsigned short* encw0 = wsb + OFF_ENC_W0;
  const unsigned short* encw1 = wsb + OFF_ENC_W1;
  const unsigned short* encw2 = wsb + OFF_ENC_W2;
  const unsigned short* dynw2 = wsb + OFF_DYN_W2;
  const unsigned short* wih   = wsb + OFF_WIH;
  const unsigned short* whh   = wsb + OFF_WHH;
  unsigned short* latb        = wsb + OFF_LAT;
  const unsigned short* dynw0 = wlds;            // LDS-resident
  const unsigned short* dynw1 = wlds + 32768;    // LDS-resident

  // ---- TDM preload of dyn_w0+dyn_w1 (contiguous 196608B) into LDS ----
  if (wave == 0) {
    tdm_load_1d(wsb + OFF_DYN_W0, wlds, WLDS_ELEMS * 2u / 8u);
    __builtin_amdgcn_s_wait_tensorcnt(0);
  }
  __syncthreads();

  // ---- encoder: OB -> H -> H -> L ----
  layer_f32A(ob + (long)b0 * OBD, OBD, OBD, encw0, enc_b0, HDIM, &h1b[0][0],
             nullptr, true, wave, lane);
  __syncthreads();
  layer_lds(&h1b[0][0], HDIM, encw1, HDIM, enc_b1, HDIM, &h2b[0][0], nullptr, true,
            wave, lane);
  __syncthreads();
  layer_lds(&h2b[0][0], HDIM, encw2, HDIM, enc_b2, LDIM, &yb[0][0], &yf[0][0], false,
            wave, lane);
  __syncthreads();

  // ---- GRU with acs[:,0] ----
  layer_f32A(acs + ((long)b0 * TDIM + 0) * ACD, TDIM * ACD, ACD, wih, gru_bih, GDIM,
             nullptr, &scr.g[0][0][0], false, wave, lane);
  layer_lds(&yb[0][0], LDIM, whh, LDIM, nullptr, GDIM, nullptr, &scr.g[1][0][0],
            false, wave, lane);
  __syncthreads();
  gru_apply(&scr.g[0][0][0], &scr.g[1][0][0], gru_bn, yf, yb, latb, 0, b0, tid);
  __syncthreads();

  // ---- sequential scan over observation intervals ----
  for (int t = 1; t < TDIM; ++t) {
    if (tid < MT) {
      float t1 = times[(b0 + tid) * TDIM + t];
      float t0 = times[(b0 + tid) * TDIM + t - 1];
      hstep[tid] = (t1 - t0) * 0.5f;  // dt / NSUB
    }
    __syncthreads();

    for (int sub = 0; sub < 2; ++sub) {
      // stage 0 uses y itself
      eval_f(&yb[0][0], dynw0, dynw1, dynw2, dyn_b0, dyn_b1, dyn_b2, h1b, h2b,
             &scr.kf[0][0][0], wave, lane);
      for (int s = 0; s < 5; ++s) {
        // stage s+1 input: y + h * sum_j A[s][j] * k_j
        for (int idx = tid; idx < MT * LDIM; idx += NTHREADS) {
          int m = idx >> 7, j = idx & (LDIM - 1);
          float acc = 0.f;
          for (int jj = 0; jj <= s; ++jj) acc += cDA[s][jj] * scr.kf[jj][m][j];
          xb[m][j] = f2bf(yf[m][j] + hstep[m] * acc);
        }
        __syncthreads();
        eval_f(&xb[0][0], dynw0, dynw1, dynw2, dyn_b0, dyn_b1, dyn_b2, h1b, h2b,
               &scr.kf[s + 1][0][0], wave, lane);
      }
      // y += h * sum_j B[j] * k_j   (B[1] == 0)
      for (int idx = tid; idx < MT * LDIM; idx += NTHREADS) {
        int m = idx >> 7, j = idx & (LDIM - 1);
        float acc = cDB[0] * scr.kf[0][m][j] + cDB[2] * scr.kf[2][m][j] +
                    cDB[3] * scr.kf[3][m][j] + cDB[4] * scr.kf[4][m][j] +
                    cDB[5] * scr.kf[5][m][j];
        float y = yf[m][j] + hstep[m] * acc;
        yf[m][j] = y;
        yb[m][j] = f2bf(y);
      }
      __syncthreads();
    }

    // GRU with acs[:,t]
    layer_f32A(acs + ((long)b0 * TDIM + t) * ACD, TDIM * ACD, ACD, wih, gru_bih,
               GDIM, nullptr, &scr.g[0][0][0], false, wave, lane);
    layer_lds(&yb[0][0], LDIM, whh, LDIM, nullptr, GDIM, nullptr, &scr.g[1][0][0],
              false, wave, lane);
    __syncthreads();
    gru_apply(&scr.g[0][0][0], &scr.g[1][0][0], gru_bn, yf, yb, latb, t, b0, tid);
    __syncthreads();
  }
}

// ---------------------------------------------------------------------------
// Decoder: fully parallel, 1024 blocks (one per 16-row latent tile).
// ---------------------------------------------------------------------------
__global__ __launch_bounds__(NTHREADS) void ode_rnn_dec(
    const unsigned short* __restrict__ latb, const unsigned short* __restrict__ w0,
    const unsigned short* __restrict__ w1, const unsigned short* __restrict__ w2,
    const float* __restrict__ b0v, const float* __restrict__ b1v,
    const float* __restrict__ b2v, float* __restrict__ out) {
  __shared__ __align__(32) unsigned short h1b[MT][HDIM];
  __shared__ __align__(32) unsigned short h2b[MT][HDIM];
  const int t = blockIdx.x >> 4;
  const int bb = (blockIdx.x & 15) * MT;
  const int tid = threadIdx.x, lane = tid & 31, wave = tid >> 5;
  const unsigned short* A = latb + ((long)t * BDIM + bb) * LDIM;

  layer_lds(A, LDIM, w0, LDIM, b0v, HDIM, &h1b[0][0], nullptr, true, wave, lane);
  __syncthreads();
  layer_lds(&h1b[0][0], HDIM, w1, HDIM, b1v, HDIM, &h2b[0][0], nullptr, true, wave,
            lane);
  __syncthreads();
  // final: N = 32 (waves 0,1), write f32 obs to out[b][t][:]
  for (int n0 = wave * 16; n0 < OBD; n0 += NWAVES * 16) {
    f32x8 acc = {};
    for (int k0 = 0; k0 < HDIM; k0 += 32)
      acc = wmma_bf16(a_frag_lds(&h2b[0][0], HDIM, k0, lane),
                      b_frag(w2, HDIM, n0, k0, lane), acc);
    int nl = lane & 15, hf = lane >> 4;
    float bv = b2v[n0 + nl];
#pragma unroll
    for (int r = 0; r < 8; ++r) {
      int m = hf * 8 + r;
      out[((long)(bb + m) * TDIM + t) * OBD + n0 + nl] = acc[r] + bv;
    }
  }
}

// ---------------------------------------------------------------------------
// Weight prep: f32 -> bf16 (raw ushort) into workspace.
// ---------------------------------------------------------------------------
__global__ void cvt_kernel(const float* __restrict__ s, unsigned short* __restrict__ d,
                           int n) {
  int i = blockIdx.x * blockDim.x + threadIdx.x;
  if (i < n) d[i] = f2bf(s[i]);
}

__global__ void cvt_pad_kernel(const float* __restrict__ s,
                               unsigned short* __restrict__ d, int rows, int kin,
                               int kout) {
  int i = blockIdx.x * blockDim.x + threadIdx.x;
  if (i < rows * kout) {
    int r = i / kout, c = i % kout;
    d[i] = (c < kin) ? f2bf(s[r * kin + c]) : (unsigned short)0;
  }
}

extern "C" void kernel_launch(void* const* d_in, const int* in_sizes, int n_in,
                              void* d_out, int out_size, void* d_ws, size_t ws_size,
                              hipStream_t stream) {
  (void)in_sizes; (void)n_in; (void)out_size; (void)ws_size;
  const float* ob     = (const float*)d_in[0];
  const float* acs    = (const float*)d_in[1];
  const float* times  = (const float*)d_in[2];
  const float* enc_b0 = (const float*)d_in[4];
  const float* enc_b1 = (const float*)d_in[6];
  const float* enc_b2 = (const float*)d_in[8];
  const float* dyn_b0 = (const float*)d_in[10];
  const float* dyn_b1 = (const float*)d_in[12];
  const float* dyn_b2 = (const float*)d_in[14];
  const float* dec_b0 = (const float*)d_in[16];
  const float* dec_b1 = (const float*)d_in[18];
  const float* dec_b2 = (const float*)d_in[20];
  const float* gru_bih = (const float*)d_in[23];
  const float* gru_bn  = (const float*)d_in[24];
  unsigned short* wsb = (unsigned short*)d_ws;

  auto cvt = [&](int idx, long off, int n) {
    cvt_kernel<<<(n + 255) / 256, 256, 0, stream>>>((const float*)d_in[idx],
                                                    wsb + off, n);
  };
  cvt(3,  OFF_ENC_W0, 256 * 32);
  cvt(5,  OFF_ENC_W1, 256 * 256);
  cvt(7,  OFF_ENC_W2, 128 * 256);
  cvt(9,  OFF_DYN_W0, 256 * 128);
  cvt(11, OFF_DYN_W1, 256 * 256);
  cvt(13, OFF_DYN_W2, 128 * 256);
  cvt(15, OFF_DEC_W0, 256 * 128);
  cvt(17, OFF_DEC_W1, 256 * 256);
  cvt(19, OFF_DEC_W2, 32 * 256);
  cvt_pad_kernel<<<(384 * 32 + 255) / 256, 256, 0, stream>>>(
      (const float*)d_in[21], wsb + OFF_WIH, 384, 16, 32);
  cvt(22, OFF_WHH, 384 * 128);

  ode_rnn_seq<<<BDIM / MT, NTHREADS, 0, stream>>>(
      ob, acs, times, wsb, enc_b0, enc_b1, enc_b2, dyn_b0, dyn_b1, dyn_b2, gru_bih,
      gru_bn);

  ode_rnn_dec<<<TDIM * (BDIM / MT), NTHREADS, 0, stream>>>(
      wsb + OFF_LAT, wsb + OFF_DEC_W0, wsb + OFF_DEC_W1, wsb + OFF_DEC_W2, dec_b0,
      dec_b1, dec_b2, (float*)d_out);
}
